// MLPPredictor_90366111908223
// MI455X (gfx1250) — matrix operand order repaired and checked by reading, not compiled, not run on hardware
//
#include <hip/hip_runtime.h>
#include <hip/hip_bf16.h>

// CDNA5 / gfx1250 WMMA types (wave32)
typedef __attribute__((ext_vector_type(16))) __bf16 v16bf_t;
typedef __attribute__((ext_vector_type(8)))  float  v8f_t;

#define GNN_N_EDGES   400000
#define GNN_D_SLICE   64
#define GNN_K_TOTAL   512          // 8 * D_SLICE
#define GNN_OUT       32
#define TILES_PER_BLK 8            // 8 waves * 16 edges = 128 edges / block

__global__ __launch_bounds__(256) void edge_mlp_wmma_kernel(
    const float* __restrict__ s0, const float* __restrict__ s1,
    const float* __restrict__ s2, const float* __restrict__ s3,
    const int*   __restrict__ src, const int* __restrict__ dst,
    const float* __restrict__ W,  const float* __restrict__ bias,
    float* __restrict__ out, int n_edges)
{
    // W staged as bf16, row-major [32][512]  -> 32 KB LDS
    __shared__ __attribute__((aligned(64))) __bf16 Wlds[GNN_OUT * GNN_K_TOTAL];

    const int tid = threadIdx.x;

    // ---- Stage W (fp32 -> bf16) cooperatively: 16384 elems, 4 at a time ----
    {
        const float4* W4 = (const float4*)W;
        #pragma unroll 4
        for (int i = tid; i < (GNN_OUT * GNN_K_TOTAL / 4); i += 256) {
            float4 w = W4[i];
            Wlds[4 * i + 0] = (__bf16)w.x;
            Wlds[4 * i + 1] = (__bf16)w.y;
            Wlds[4 * i + 2] = (__bf16)w.z;
            Wlds[4 * i + 3] = (__bf16)w.w;
        }
    }
    __syncthreads();

    const int lane  = tid & 31;
    const int wid   = tid >> 5;
    const int tile  = blockIdx.x * TILES_PER_BLK + wid;   // 16 edges per tile
    if (tile * 16 >= n_edges) return;                     // wave-uniform guard

    const int m     = lane & 15;   // A-row (edge within tile) / B-col (class) / C-col N
    const int khalf = lane >> 4;   // which K half-window this lane covers

    const int e     = tile * 16 + m;
    const int nsrc  = src[e];
    const int ndst  = dst[e];
    const float* const tbl_src[4] = { s0 + (size_t)nsrc * GNN_D_SLICE,
                                      s1 + (size_t)nsrc * GNN_D_SLICE,
                                      s2 + (size_t)nsrc * GNN_D_SLICE,
                                      s3 + (size_t)nsrc * GNN_D_SLICE };
    const float* const tbl_dst[4] = { s0 + (size_t)ndst * GNN_D_SLICE,
                                      s1 + (size_t)ndst * GNN_D_SLICE,
                                      s2 + (size_t)ndst * GNN_D_SLICE,
                                      s3 + (size_t)ndst * GNN_D_SLICE };

    // Bias folded into accumulator init. C layout: N = lane%16 (+16 for tile1),
    // identical across all 8 C VGPRs.
    const float b0 = bias[m];
    const float b1 = bias[16 + m];
    v8f_t acc0, acc1;
    #pragma unroll
    for (int r = 0; r < 8; ++r) { acc0[r] = b0; acc1[r] = b1; }

    // B fragments: lane c holds 16 contiguous K values of class c (bf16), in the contiguous
    // K order A also uses (lanes 0-15: K=chunk+0..15, lanes 16-31: K=chunk+16..31).
    const __bf16* wrow0 = &Wlds[(size_t)(m)      * GNN_K_TOTAL];
    const __bf16* wrow1 = &Wlds[(size_t)(16 + m) * GNN_K_TOTAL];

    #pragma unroll
    for (int kc = 0; kc < 16; ++kc) {
        // K-chunk [kc*32, kc*32+32) lies inside slice (kc>>1); slices 0-3 = src side,
        // 4-7 = dst side. Fully unrolled -> table/node resolve at compile time.
        const int slice = kc >> 1;
        const float* base = ((slice < 4) ? tbl_src[slice & 3] : tbl_dst[slice & 3])
                            + ((kc & 1) * 32) + khalf * 16;

        // A fragment gather: 16 contiguous floats (K+0..15 for this lane half)
        float4 a0 = *(const float4*)(base + 0);
        float4 a1 = *(const float4*)(base + 4);
        float4 a2 = *(const float4*)(base + 8);
        float4 a3 = *(const float4*)(base + 12);

        v16bf_t A;
        A[0]  = (__bf16)a0.x; A[1]  = (__bf16)a0.y; A[2]  = (__bf16)a0.z; A[3]  = (__bf16)a0.w;
        A[4]  = (__bf16)a1.x; A[5]  = (__bf16)a1.y; A[6]  = (__bf16)a1.z; A[7]  = (__bf16)a1.w;
        A[8]  = (__bf16)a2.x; A[9]  = (__bf16)a2.y; A[10] = (__bf16)a2.z; A[11] = (__bf16)a2.w;
        A[12] = (__bf16)a3.x; A[13] = (__bf16)a3.y; A[14] = (__bf16)a3.z; A[15] = (__bf16)a3.w;

        const int kb = kc * 32 + khalf * 16;        // 32B-aligned in bf16 units
        v16bf_t B0 = *(const v16bf_t*)(wrow0 + kb);
        v16bf_t B1 = *(const v16bf_t*)(wrow1 + kb);

        acc0 = __builtin_amdgcn_wmma_f32_16x16x32_bf16(false, A, false, B0,
                                                       (short)0, acc0, false, false);
        acc1 = __builtin_amdgcn_wmma_f32_16x16x32_bf16(false, A, false, B1,
                                                       (short)0, acc1, false, false);
    }

    // Store: element (M = r + 8*khalf, N = m + 16*t) -> out[(tile*16+M)*32 + N].
    // Per r, lanes form two contiguous 64B spans -> coalesced.
    const size_t obase = (size_t)tile * 16 * GNN_OUT;
    #pragma unroll
    for (int r = 0; r < 8; ++r) {
        const size_t row = obase + (size_t)(r + 8 * khalf) * GNN_OUT;
        out[row + m]      = acc0[r];
        out[row + 16 + m] = acc1[r];
    }
}

extern "C" void kernel_launch(void* const* d_in, const int* in_sizes, int n_in,
                              void* d_out, int out_size, void* d_ws, size_t ws_size,
                              hipStream_t stream) {
    const float* s0  = (const float*)d_in[0];
    const float* s1  = (const float*)d_in[1];
    const float* s2  = (const float*)d_in[2];
    const float* s3  = (const float*)d_in[3];
    const int*   src = (const int*)d_in[4];
    const int*   dst = (const int*)d_in[5];
    const float* W   = (const float*)d_in[6];
    const float* b   = (const float*)d_in[7];
    float* out = (float*)d_out;

    const int n_edges = in_sizes[4];                       // 400000
    const int tiles   = (n_edges + 15) / 16;               // 25000
    const int blocks  = (tiles + TILES_PER_BLK - 1) / TILES_PER_BLK; // 3125

    edge_mlp_wmma_kernel<<<blocks, 256, 0, stream>>>(
        s0, s1, s2, s3, src, dst, W, b, out, n_edges);
}
